// SpeakerEmbedderGeeArrYou_16295105921445
// MI455X (gfx1250) — compile-verified, matrix operand order checked by reading, not run on hardware
//
#include <hip/hip_runtime.h>
#include <hip/hip_bf16.h>
#include <math.h>

typedef __bf16 bf16;
typedef __bf16 v16bf __attribute__((ext_vector_type(16)));
typedef float  v8f   __attribute__((ext_vector_type(8)));

#define B_   128
#define T_   400
#define M_   80
#define MP_  96          // M padded to 3 K-tiles of 32
#define H_   768
#define G_   2304        // 3*H
#define FC_  256
#define KT_H 24          // 768/32
#define NBLK 144         // 3 layers * 48 column tiles (persistent grid)
// dynamic LDS: worst case (layers 1/2) = (3*KT_H + 3*KT_H) * 512 bf16 elems
#define LDS_BYTES ((3 * KT_H + 3 * KT_H) * 512 * 2)

// K offset within a 32-wide K-tile for bf16 A/B fragment element pair j, lane-half h.
// (16-bit A-matrix 16x32 layout, CDNA5 ISA §7.12.2; B mirrors with lanes = columns.)
__device__ __forceinline__ int kOff(int half, int j) {
    return (j < 4) ? (2 * j + half * 8) : (16 + 2 * (j - 4) + half * 8);
}

// ---------------- pre-pass: convert x to bf16, pad 80 -> 96 -------------------
__global__ __launch_bounds__(256) void pad_x_kernel(const float* __restrict__ x,
                                                    bf16* __restrict__ xp) {
    int idx = blockIdx.x * blockDim.x + threadIdx.x;
    if (idx >= B_ * T_ * MP_) return;
    int k  = idx % MP_;
    int bt = idx / MP_;
    float v = (k < M_) ? x[(size_t)bt * M_ + k] : 0.0f;
    xp[idx] = (bf16)v;
}

// ---- pre-pass: swizzle weight matrix W[N x Ksrc] (used as B = W^T) into
// WMMA-fragment-ready bf16 layout (one coalesced 32B/lane load per 32x16 tile).
__global__ __launch_bounds__(256) void swizzle_w_kernel(const float* __restrict__ W,
                                                        bf16* __restrict__ F,
                                                        int Ksrc, int KT, int nTiles) {
    int idx = blockIdx.x * blockDim.x + threadIdx.x;
    int total = nTiles * KT * 512;
    if (idx >= total) return;
    int e    = idx & 15;
    int lane = (idx >> 4) & 31;
    int rest = idx >> 9;
    int kt   = rest % KT;
    int nt   = rest / KT;
    int half = lane >> 4;
    int j    = e >> 1;
    int pair = e & 1;
    int k = kt * 32 + kOff(half, j) + pair;
    int n = nt * 16 + (lane & 15);
    float w = (k < Ksrc) ? W[(size_t)n * Ksrc + k] : 0.0f;
    F[idx] = (bf16)w;
}

// ---------------- persistent-scan argument block ------------------------------
struct ScanArgs {
    const bf16* xpad;
    const bf16* wih[3];     // fragment-swizzled W_ih per layer
    const bf16* whh[3];     // fragment-swizzled W_hh per layer
    const float* bih[3];
    const float* bhh[3];
    float* hf[3][2];        // f32 h state, ping-pong per layer
    bf16*  hb[3][2];        // bf16 h state, ping-pong per layer
    bf16*  hand[2][2];      // layer->layer handoff, double buffered by t parity
    unsigned* barCnt;
    unsigned* barGen;
};

// ---------------- init: zero states + barrier (every call, for determinism) ---
__global__ __launch_bounds__(256) void init_state_kernel(ScanArgs a) {
    int idx = blockIdx.x * blockDim.x + threadIdx.x;
    if (idx == 0) { *a.barCnt = 0u; *a.barGen = 0u; }
    if (idx < B_ * H_) {
        #pragma unroll
        for (int l = 0; l < 3; ++l) {
            #pragma unroll
            for (int p = 0; p < 2; ++p) {
                a.hf[l][p][idx] = 0.0f;
                a.hb[l][p][idx] = (bf16)0.0f;
            }
        }
        #pragma unroll
        for (int l = 0; l < 2; ++l) {
            #pragma unroll
            for (int p = 0; p < 2; ++p)
                a.hand[l][p][idx] = (bf16)0.0f;
        }
    }
}

// ---------------- device-wide sense-reversing barrier -------------------------
__device__ __forceinline__ void grid_sync(unsigned* cnt, unsigned* gen) {
    __syncthreads();
    if (threadIdx.x == 0) {
        __threadfence();   // release this block's writes to device scope
        unsigned g = __hip_atomic_load(gen, __ATOMIC_RELAXED, __HIP_MEMORY_SCOPE_AGENT);
        unsigned old = __hip_atomic_fetch_add(cnt, 1u, __ATOMIC_ACQ_REL, __HIP_MEMORY_SCOPE_AGENT);
        if (old == NBLK - 1u) {
            __hip_atomic_store(cnt, 0u, __ATOMIC_RELAXED, __HIP_MEMORY_SCOPE_AGENT);
            __hip_atomic_fetch_add(gen, 1u, __ATOMIC_RELEASE, __HIP_MEMORY_SCOPE_AGENT);
        } else {
            while (__hip_atomic_load(gen, __ATOMIC_ACQUIRE, __HIP_MEMORY_SCOPE_AGENT) == g)
                __builtin_amdgcn_s_sleep(2);
        }
        __threadfence();   // acquire other blocks' writes
    }
    __syncthreads();
}

// ---- one [16 rows] x [3 gates x 16 cols] x K GEMM chunk; B tiles from LDS ----
// sW layout: gate-major, ((g*KT + kt)*32 + lane)*16 bf16 elements.
__device__ __forceinline__ void gemm_gates_lds(const bf16* __restrict__ aBase, int KT,
                                               const bf16* sW,
                                               int lane, int half,
                                               v8f& accR, v8f& accZ, v8f& accN) {
    const int gateStride = KT * 512;
    for (int kt = 0; kt < KT; ++kt) {
        union { v16bf v; unsigned u[8]; } aa;
        const bf16* ap = aBase + kt * 32;
        #pragma unroll
        for (int j = 0; j < 8; ++j)
            aa.u[j] = *(const unsigned*)(ap + kOff(half, j));
        const bf16* sB = sW + ((size_t)kt * 32 + lane) * 16;
        const v16bf bR = *(const v16bf*)(sB);
        const v16bf bZ = *(const v16bf*)(sB + gateStride);
        const v16bf bN = *(const v16bf*)(sB + 2 * gateStride);
        accR = __builtin_amdgcn_wmma_f32_16x16x32_bf16(false, aa.v, false, bR, (short)0, accR, false, false);
        accZ = __builtin_amdgcn_wmma_f32_16x16x32_bf16(false, aa.v, false, bZ, (short)0, accZ, false, false);
        accN = __builtin_amdgcn_wmma_f32_16x16x32_bf16(false, aa.v, false, bN, (short)0, accN, false, false);
    }
}

// ---------------- persistent GRU scan: 3 layers wavefront-pipelined -----------
// grid = 144 blocks (layer = blk/48, column tile = blk%48), block = 8 waves.
// Weights for this block's 16 columns staged in LDS once, reused for 400 steps.
__global__ __launch_bounds__(256) void gru_scan_kernel(ScanArgs a) {
    extern __shared__ __align__(16) char smemRaw[];
    bf16* smem = (bf16*)smemRaw;

    const int layer = blockIdx.x / 48;
    const int cTile = blockIdx.x % 48;
    const int lane  = threadIdx.x & 31;
    const int wave  = threadIdx.x >> 5;
    const int half  = lane >> 4;
    const int rowA  = lane & 15;          // A-fragment row within M-tile
    const int mBase = wave * 16;
    const int col   = cTile * 16 + (lane & 15);

    const int KTin = (layer == 0) ? (MP_ >> 5) : KT_H;

    // LDS layout: [ih: 3*KTin*512][hh: 3*KT_H*512]
    bf16* sIh = smem;
    bf16* sHh = smem + 3 * KTin * 512;

    // ---- stage this block's weight slice into LDS (once) ----
    {
        const int tid = threadIdx.x;
        #pragma unroll
        for (int g = 0; g < 3; ++g) {
            const size_t nt = (size_t)g * 48 + cTile;
            const uint4* src = (const uint4*)(a.wih[layer] + nt * KTin * 512);
            uint4* dst = (uint4*)(sIh + (size_t)g * KTin * 512);
            const int chunks = KTin * 64;   // 16B chunks
            for (int i = tid; i < chunks; i += 256) dst[i] = src[i];
        }
        #pragma unroll
        for (int g = 0; g < 3; ++g) {
            const size_t nt = (size_t)g * 48 + cTile;
            const uint4* src = (const uint4*)(a.whh[layer] + nt * KT_H * 512);
            uint4* dst = (uint4*)(sHh + (size_t)g * KT_H * 512);
            const int chunks = KT_H * 64;
            for (int i = tid; i < chunks; i += 256) dst[i] = src[i];
        }
        __syncthreads();
    }

    const float biR = a.bih[layer][col],          bhR = a.bhh[layer][col];
    const float biZ = a.bih[layer][H_ + col],     bhZ = a.bhh[layer][H_ + col];
    const float biN = a.bih[layer][2 * H_ + col], bhN = a.bhh[layer][2 * H_ + col];

    for (int s = 0; s < T_ + 2; ++s) {
        const int t = s - layer;
        if (t >= 0 && t < T_) {
            const int p = t & 1;
            v8f accR = {}; v8f accZ = {}; v8f accHN = {}; v8f accIN = {};

            // input contribution: x_t @ W_ih^T  (layer 0 from x, else from handoff)
            const bf16* inBase = (layer == 0)
                ? a.xpad + ((size_t)(mBase + rowA) * T_ + t) * MP_
                : a.hand[layer - 1][p] + (size_t)(mBase + rowA) * H_;
            gemm_gates_lds(inBase, KTin, sIh, lane, half, accR, accZ, accIN);

            // recurrent contribution: h_{t-1} @ W_hh^T
            const bf16* hBase = a.hb[layer][p] + (size_t)(mBase + rowA) * H_;
            gemm_gates_lds(hBase, KT_H, sHh, lane, half, accR, accZ, accHN);

            // gate math + state update (f32 master state)
            const float* hfIn  = a.hf[layer][p];
            float*       hfOut = a.hf[layer][1 - p];
            bf16*        hbOut = a.hb[layer][1 - p];
            bf16*        hndOut = (layer < 2) ? a.hand[layer][p] : (bf16*)0;
            #pragma unroll
            for (int v = 0; v < 8; ++v) {
                int row = mBase + v + (half << 3);  // C/D layout: VGPR v -> M=v / v+8
                float hold = hfIn[(size_t)row * H_ + col];
                float r = 1.0f / (1.0f + __expf(-(accR[v] + biR + bhR)));
                float z = 1.0f / (1.0f + __expf(-(accZ[v] + biZ + bhZ)));
                float n = tanhf(accIN[v] + biN + r * (accHN[v] + bhN));
                float hn = (1.0f - z) * n + z * hold;
                hfOut[(size_t)row * H_ + col] = hn;
                bf16 hbv = (bf16)hn;
                hbOut[(size_t)row * H_ + col] = hbv;
                if (layer < 2) hndOut[(size_t)row * H_ + col] = hbv;
            }
        }
        grid_sync(a.barCnt, a.barGen);
    }
}

// ---------------- FC: out = h_last @ W_fc^T + b_fc  (WMMA) --------------------
__global__ __launch_bounds__(256) void fc_gemm_kernel(
    const bf16* __restrict__ hlast, const bf16* __restrict__ wfcF,
    const float* __restrict__ b_fc, float* __restrict__ out)
{
    const int lane  = threadIdx.x & 31;
    const int wave  = threadIdx.x >> 5;
    const int half  = lane >> 4;
    const int rowA  = lane & 15;
    const int mBase = wave * 16;
    const size_t nTile = blockIdx.x;
    const int colBase = (int)nTile * 16;

    v8f acc = {};
    const bf16* aRow = hlast + (size_t)(mBase + rowA) * H_;
    for (int kt = 0; kt < KT_H; ++kt) {
        union { v16bf v; unsigned u[8]; } aa;
        const bf16* ap = aRow + kt * 32;
        #pragma unroll
        for (int j = 0; j < 8; ++j)
            aa.u[j] = *(const unsigned*)(ap + kOff(half, j));
        const v16bf b = *(const v16bf*)(wfcF + ((nTile * KT_H + kt) * 32 + lane) * 16);
        acc = __builtin_amdgcn_wmma_f32_16x16x32_bf16(false, aa.v, false, b, (short)0, acc, false, false);
    }
    const int col = colBase + (lane & 15);
    const float bias = b_fc[col];
    #pragma unroll
    for (int v = 0; v < 8; ++v) {
        int row = mBase + v + (half << 3);
        out[(size_t)row * FC_ + col] = acc[v] + bias;
    }
}

// ---------------- L2 normalize rows of [128 x 256] ----------------------------
__global__ __launch_bounds__(256) void l2norm_kernel(const float* __restrict__ in,
                                                     float* __restrict__ out) {
    __shared__ float red[8];
    const int row = blockIdx.x;
    float v = in[(size_t)row * FC_ + threadIdx.x];
    float s = v * v;
    #pragma unroll
    for (int o = 16; o > 0; o >>= 1) s += __shfl_down(s, o, 32);
    if ((threadIdx.x & 31) == 0) red[threadIdx.x >> 5] = s;
    __syncthreads();
    if (threadIdx.x == 0) {
        float tot = 0.0f;
        #pragma unroll
        for (int i = 0; i < 8; ++i) tot += red[i];
        red[0] = tot;
    }
    __syncthreads();
    out[(size_t)row * FC_ + threadIdx.x] = v * rsqrtf(red[0]);
}

// ------------------------------- host side ------------------------------------
extern "C" void kernel_launch(void* const* d_in, const int* in_sizes, int n_in,
                              void* d_out, int out_size, void* d_ws, size_t ws_size,
                              hipStream_t stream) {
    (void)in_sizes; (void)n_in; (void)out_size; (void)ws_size;
    const float* x      = (const float*)d_in[0];
    const float* W_ih0  = (const float*)d_in[1];
    const float* W_hh0  = (const float*)d_in[2];
    const float* b_ih0  = (const float*)d_in[3];
    const float* b_hh0  = (const float*)d_in[4];
    const float* W_ih_r = (const float*)d_in[5];   // [2, 2304, 768]
    const float* W_hh_r = (const float*)d_in[6];   // [2, 2304, 768]
    const float* b_ih_r = (const float*)d_in[7];   // [2, 2304]
    const float* b_hh_r = (const float*)d_in[8];   // [2, 2304]
    const float* W_fc   = (const float*)d_in[9];
    const float* b_fc   = (const float*)d_in[10];

    // ---- carve workspace ----
    char* ws = (char*)d_ws;
    size_t off = 0;
    auto carve = [&](size_t bytes) -> char* {
        char* p = ws + off;
        off = (off + bytes + 255) & ~(size_t)255;
        return p;
    };
    bf16*  xpad   = (bf16*)carve((size_t)B_ * T_ * MP_ * 2);
    bf16*  wih0F  = (bf16*)carve((size_t)144 * 3 * 512 * 2);
    bf16*  wihF1  = (bf16*)carve((size_t)144 * KT_H * 512 * 2);
    bf16*  wihF2  = (bf16*)carve((size_t)144 * KT_H * 512 * 2);
    bf16*  whhF0  = (bf16*)carve((size_t)144 * KT_H * 512 * 2);
    bf16*  whhF1  = (bf16*)carve((size_t)144 * KT_H * 512 * 2);
    bf16*  whhF2  = (bf16*)carve((size_t)144 * KT_H * 512 * 2);
    bf16*  wfcF   = (bf16*)carve((size_t)16 * KT_H * 512 * 2);
    float* fc_tmp = (float*)carve((size_t)B_ * FC_ * 4);
    unsigned* bar = (unsigned*)carve(256);

    ScanArgs a;
    a.xpad = xpad;
    a.wih[0] = wih0F; a.wih[1] = wihF1; a.wih[2] = wihF2;
    a.whh[0] = whhF0; a.whh[1] = whhF1; a.whh[2] = whhF2;
    a.bih[0] = b_ih0; a.bih[1] = b_ih_r;           a.bih[2] = b_ih_r + G_;
    a.bhh[0] = b_hh0; a.bhh[1] = b_hh_r;           a.bhh[2] = b_hh_r + G_;
    for (int l = 0; l < 3; ++l)
        for (int p = 0; p < 2; ++p) {
            a.hf[l][p] = (float*)carve((size_t)B_ * H_ * 4);
            a.hb[l][p] = (bf16*)carve((size_t)B_ * H_ * 2);
        }
    for (int l = 0; l < 2; ++l)
        for (int p = 0; p < 2; ++p)
            a.hand[l][p] = (bf16*)carve((size_t)B_ * H_ * 2);
    a.barCnt = bar;
    a.barGen = bar + 16;

    // ---- pre-passes: bf16 conversion + fragment swizzle ----
    {
        int n = B_ * T_ * MP_;
        pad_x_kernel<<<(n + 255) / 256, 256, 0, stream>>>(x, xpad);
    }
    auto swz = [&](const float* W, bf16* F, int Ksrc, int KT, int nTiles) {
        int total = nTiles * KT * 512;
        swizzle_w_kernel<<<(total + 255) / 256, 256, 0, stream>>>(W, F, Ksrc, KT, nTiles);
    };
    swz(W_ih0, wih0F, M_, 3, 144);
    swz(W_ih_r + 0 * (size_t)G_ * H_, wihF1, H_, KT_H, 144);
    swz(W_ih_r + 1 * (size_t)G_ * H_, wihF2, H_, KT_H, 144);
    swz(W_hh0, whhF0, H_, KT_H, 144);
    swz(W_hh_r + 0 * (size_t)G_ * H_, whhF1, H_, KT_H, 144);
    swz(W_hh_r + 1 * (size_t)G_ * H_, whhF2, H_, KT_H, 144);
    swz(W_fc, wfcF, H_, KT_H, 16);

    // ---- init states + barrier, then one persistent scan kernel ----
    {
        int n = B_ * H_;
        init_state_kernel<<<(n + 255) / 256, 256, 0, stream>>>(a);
    }
    gru_scan_kernel<<<NBLK, 256, LDS_BYTES, stream>>>(a);

    // final h of layer 2: t=399 wrote parity buffer 1-(399&1) = 0
    fc_gemm_kernel<<<16, 256, 0, stream>>>(a.hb[2][0], wfcF, b_fc, fc_tmp);
    l2norm_kernel<<<B_, 256, 0, stream>>>(fc_tmp, (float*)d_out);
}
